// MambaBlock_23502061044157
// MI455X (gfx1250) — matrix operand order repaired and checked by reading, not compiled, not run on hardware
//
#include <hip/hip_runtime.h>

#define BB 2
#define LL 2048
#define DM 1024
#define DI 2048
#define DSN 16
#define DR 64
#define NROW (BB * LL)          // 4096 rows (B*L)
#define NXZ  (2 * DI)           // 4096
#define NDBC 96                 // DT_RANK + 2*D_STATE

typedef __attribute__((ext_vector_type(16))) __bf16 v16bf;
typedef __attribute__((ext_vector_type(8)))  float  v8f;

union Frag16 { v16bf v; unsigned int u[8]; };

__device__ __forceinline__ unsigned short f32_to_bf16(float f) {
    unsigned int u = __float_as_uint(f);
    unsigned int r = 0x7FFFu + ((u >> 16) & 1u);
    return (unsigned short)((u + r) >> 16);
}
__device__ __forceinline__ float bf16_to_f32(unsigned short h) {
    return __uint_as_float(((unsigned int)h) << 16);
}
__device__ __forceinline__ float sigmoid_f(float x) {
    return 1.0f / (1.0f + __expf(-x));
}

// ---------------- LayerNorm: one block per row, bf16 output ----------------
__global__ void k_layernorm(const float* __restrict__ x,
                            const float* __restrict__ g,
                            const float* __restrict__ b,
                            unsigned short* __restrict__ xn) {
    __shared__ float red[256];
    const int row = blockIdx.x;
    const int tid = threadIdx.x;
    const float* xr = x + (size_t)row * DM;

    float s = 0.f;
    for (int i = tid; i < DM; i += 256) s += xr[i];
    red[tid] = s; __syncthreads();
    for (int off = 128; off > 0; off >>= 1) {
        if (tid < off) red[tid] += red[tid + off];
        __syncthreads();
    }
    const float mu = red[0] * (1.0f / DM);
    __syncthreads();

    float vsum = 0.f;
    for (int i = tid; i < DM; i += 256) { float d = xr[i] - mu; vsum += d * d; }
    red[tid] = vsum; __syncthreads();
    for (int off = 128; off > 0; off >>= 1) {
        if (tid < off) red[tid] += red[tid + off];
        __syncthreads();
    }
    const float rstd = rsqrtf(red[0] * (1.0f / DM) + 1e-5f);

    for (int i = tid; i < DM; i += 256)
        xn[(size_t)row * DM + i] = f32_to_bf16((xr[i] - mu) * rstd * g[i] + b[i]);
}

// ---------- transpose f32[R][C] -> bf16[C][R] (weights, one-time) ----------
__global__ void k_transpose_bf16(const float* __restrict__ in,
                                 unsigned short* __restrict__ out,
                                 int R, int C) {
    long long idx = (long long)blockIdx.x * 256 + threadIdx.x;
    if (idx >= (long long)R * C) return;
    int r = (int)(idx / C);
    int c = (int)(idx % C);
    out[(size_t)c * R + r] = f32_to_bf16(in[idx]);
}

// ------------- epilogue helper ---------------------------------------------
// MODE 0: split bf16 store (col<split -> Cb0, else Cb1), ldc = split width
// MODE 1: f32 store
// MODE 2: f32 store of softplus(c + bias[col])
// MODE 3: f32 store of c + resid[row*ldc+col]   (residual add)
template <int MODE>
__device__ __forceinline__ void gemm_epilogue(const v8f& c, int row_base, int col,
                                              float* __restrict__ Cf,
                                              unsigned short* __restrict__ Cb0,
                                              unsigned short* __restrict__ Cb1,
                                              int split,
                                              const float* __restrict__ extra,
                                              int ldc) {
#pragma unroll
    for (int v = 0; v < 8; ++v) {
        const int row = row_base + v;           // C/D: M = half*8 + v
        const float val = c[v];
        if (MODE == 0) {
            if (col < split) Cb0[(size_t)row * ldc + col] = f32_to_bf16(val);
            else             Cb1[(size_t)row * ldc + (col - split)] = f32_to_bf16(val);
        } else if (MODE == 1) {
            Cf[(size_t)row * ldc + col] = val;
        } else if (MODE == 2) {
            const float xx = val + extra[col];
            Cf[(size_t)row * ldc + col] = (xx > 20.f) ? xx : __logf(1.f + __expf(xx));
        } else {
            Cf[(size_t)row * ldc + col] = val + extra[(size_t)row * ldc + col];
        }
    }
}

// ---------------- WMMA GEMM: C[M,N] = A[M,K](bf16) * Bt[N,K](bf16) ----------
// One wave computes a 32x32 macro-tile: 2 A frags x 2 B frags -> 4 independent
// WMMAs per K-step (hides WMMA->WMMA RAW hazard, 2x FLOP/byte vs 1 tile).
template <int MODE>
__global__ void k_gemm_wmma(const unsigned short* __restrict__ A, int lda,
                            const unsigned short* __restrict__ Bt, int ldb,
                            int K,
                            float* __restrict__ Cf,
                            unsigned short* __restrict__ Cb0,
                            unsigned short* __restrict__ Cb1, int split,
                            const float* __restrict__ extra, int ldc) {
    const int lane = threadIdx.x;           // 0..31, wave32
    const int half = lane >> 4;             // 0 | 1
    const int mn   = lane & 15;
    const int tile_n = blockIdx.x * 2;                       // two N tiles
    const int tile_m = (blockIdx.y * 4 + threadIdx.y) * 2;   // two M tiles

    const unsigned short* Arow0 = A  + (size_t)(tile_m * 16 + mn) * lda;
    const unsigned short* Arow1 = Arow0 + (size_t)16 * lda;
    const unsigned short* Brow0 = Bt + (size_t)(tile_n * 16 + mn) * ldb;
    const unsigned short* Brow1 = Brow0 + (size_t)16 * ldb;

    v8f c00 = {}, c01 = {}, c10 = {}, c11 = {};
    for (int k0 = 0; k0 < K; k0 += 32) {
        Frag16 a0, a1, b0, b1;
#pragma unroll
        for (int v = 0; v < 8; ++v) {
            // A 16x32 bf16 layout: K = (v&4?16:0) + half*8 + 2*(v&3)
            const int ka = ((v & 4) << 2) + (half << 3) + ((v & 3) << 1);
            a0.u[v] = *reinterpret_cast<const unsigned int*>(Arow0 + k0 + ka);
            a1.u[v] = *reinterpret_cast<const unsigned int*>(Arow1 + k0 + ka);
        }
#pragma unroll
        for (int v = 0; v < 8; ++v) {
            // B 32x16 bf16 layout (col-major source): K = (v&4?16:0) + half*8 + 2*(v&3)
            const int kb = ((v & 4) << 2) + (half << 3) + ((v & 3) << 1);
            b0.u[v] = *reinterpret_cast<const unsigned int*>(Brow0 + k0 + kb);
            b1.u[v] = *reinterpret_cast<const unsigned int*>(Brow1 + k0 + kb);
        }
        if (k0 + 32 < K) {
            __builtin_prefetch(Arow0 + k0 + 32, 0, 3);  // global_prefetch_b8
            __builtin_prefetch(Arow1 + k0 + 32, 0, 3);
            __builtin_prefetch(Brow0 + k0 + 32, 0, 3);
            __builtin_prefetch(Brow1 + k0 + 32, 0, 3);
        }
        c00 = __builtin_amdgcn_wmma_f32_16x16x32_bf16(false, a0.v, false, b0.v,
                                                      (short)0, c00, false, false);
        c01 = __builtin_amdgcn_wmma_f32_16x16x32_bf16(false, a0.v, false, b1.v,
                                                      (short)0, c01, false, false);
        c10 = __builtin_amdgcn_wmma_f32_16x16x32_bf16(false, a1.v, false, b0.v,
                                                      (short)0, c10, false, false);
        c11 = __builtin_amdgcn_wmma_f32_16x16x32_bf16(false, a1.v, false, b1.v,
                                                      (short)0, c11, false, false);
    }

    const int row0 = tile_m * 16 + (half << 3);
    const int row1 = row0 + 16;
    const int col0 = tile_n * 16 + mn;
    const int col1 = col0 + 16;
    gemm_epilogue<MODE>(c00, row0, col0, Cf, Cb0, Cb1, split, extra, ldc);
    gemm_epilogue<MODE>(c01, row0, col1, Cf, Cb0, Cb1, split, extra, ldc);
    gemm_epilogue<MODE>(c10, row1, col0, Cf, Cb0, Cb1, split, extra, ldc);
    gemm_epilogue<MODE>(c11, row1, col1, Cf, Cb0, Cb1, split, extra, ldc);
}

// ------------- causal depthwise conv (k=4) + bias + SiLU -> u (bf16) -------
__global__ void k_conv_silu(const unsigned short* __restrict__ xb,
                            const float* __restrict__ w,
                            const float* __restrict__ cb,
                            unsigned short* __restrict__ u) {
    const long long idx = (long long)blockIdx.x * 256 + threadIdx.x;
    if (idx >= (long long)NROW * DI) return;
    const int d = (int)(idx % DI);
    const long long bl = idx / DI;
    const int l = (int)(bl % LL);
    const int b = (int)(bl / LL);

    float acc = cb[d];
#pragma unroll
    for (int j = 0; j < 4; ++j) {
        const int li = l - 3 + j;
        if (li >= 0)
            acc += w[d * 4 + j] * bf16_to_f32(xb[((size_t)b * LL + li) * DI + d]);
    }
    u[idx] = f32_to_bf16(acc * sigmoid_f(acc));
}

// ------------- extract dt-rank columns of dbc -> bf16 ----------------------
__global__ void k_dtp(const float* __restrict__ dbc, unsigned short* __restrict__ dtp) {
    const int idx = blockIdx.x * 256 + threadIdx.x;
    if (idx >= NROW * DR) return;
    const int r = idx / DR, c = idx % DR;
    dtp[idx] = f32_to_bf16(dbc[(size_t)r * NDBC + c]);
}

// ------------- selective scan: one lane per (b, d) -------------------------
__global__ void k_scan(const float* __restrict__ dtf,      // [NROW][DI]
                       const float* __restrict__ dbc,      // [NROW][96]
                       const unsigned short* __restrict__ u,   // [NROW][DI]
                       const unsigned short* __restrict__ z,   // [NROW][DI]
                       const float* __restrict__ A_log,    // [DI][16]
                       const float* __restrict__ Dp,       // [DI]
                       unsigned short* __restrict__ ym) {  // [NROW][DI]
    const int tid = blockIdx.x * 256 + threadIdx.x;
    if (tid >= BB * DI) return;
    const int b = tid / DI;
    const int d = tid % DI;

    float As[DSN], h[DSN];
#pragma unroll
    for (int s = 0; s < DSN; ++s) {
        As[s] = -__expf(A_log[(size_t)d * DSN + s]);
        h[s] = 0.f;
    }
    const float dpar = Dp[d];

    for (int t = 0; t < LL; ++t) {
        const size_t base = (size_t)b * LL + t;
        const float dtv = dtf[base * DI + d];
        const float uv  = bf16_to_f32(u[base * DI + d]);
        const float zv  = bf16_to_f32(z[base * DI + d]);
        const float duv = dtv * uv;
        const float* bc = dbc + base * NDBC;
        float y = 0.f;
#pragma unroll
        for (int s = 0; s < DSN; ++s) {
            const float Bv = bc[DR + s];
            const float Cv = bc[DR + DSN + s];
            h[s] = __expf(dtv * As[s]) * h[s] + duv * Bv;
            y += h[s] * Cv;
        }
        const float out = (y + uv * dpar) * (zv * sigmoid_f(zv));
        ym[base * DI + d] = f32_to_bf16(out);
    }
}

// --------------------------- workspace layout ------------------------------
#define ALIGN256(x) (((x) + 255) & ~(size_t)255)
static const size_t OFF_XN    = 0;                                              // NROW*DM bf16
static const size_t OFF_WTIN  = OFF_XN    + ALIGN256((size_t)NROW * DM * 2);    // NXZ*DM bf16
static const size_t OFF_XB    = OFF_WTIN  + ALIGN256((size_t)NXZ * DM * 2);     // NROW*DI bf16
static const size_t OFF_Z     = OFF_XB    + ALIGN256((size_t)NROW * DI * 2);
static const size_t OFF_U     = OFF_Z     + ALIGN256((size_t)NROW * DI * 2);
static const size_t OFF_WTX   = OFF_U     + ALIGN256((size_t)NROW * DI * 2);    // 96*DI bf16
static const size_t OFF_DBC   = OFF_WTX   + ALIGN256((size_t)NDBC * DI * 2);    // NROW*96 f32
static const size_t OFF_DTP   = OFF_DBC   + ALIGN256((size_t)NROW * NDBC * 4);  // NROW*64 bf16
static const size_t OFF_WTDT  = OFF_DTP   + ALIGN256((size_t)NROW * DR * 2);    // DI*64 bf16
static const size_t OFF_DTF   = OFF_WTDT  + ALIGN256((size_t)DI * DR * 2);      // NROW*DI f32
static const size_t OFF_YM    = OFF_DTF   + ALIGN256((size_t)NROW * DI * 4);    // NROW*DI bf16
static const size_t OFF_WTOUT = OFF_YM    + ALIGN256((size_t)NROW * DI * 2);    // DM*DI bf16

extern "C" void kernel_launch(void* const* d_in, const int* in_sizes, int n_in,
                              void* d_out, int out_size, void* d_ws, size_t ws_size,
                              hipStream_t stream) {
    const float* x      = (const float*)d_in[0];
    const float* ln_g   = (const float*)d_in[1];
    const float* ln_b   = (const float*)d_in[2];
    const float* W_in   = (const float*)d_in[3];
    const float* conv_w = (const float*)d_in[4];
    const float* conv_b = (const float*)d_in[5];
    const float* W_x    = (const float*)d_in[6];
    const float* W_dt   = (const float*)d_in[7];
    const float* b_dt   = (const float*)d_in[8];
    const float* A_log  = (const float*)d_in[9];
    const float* D_par  = (const float*)d_in[10];
    const float* W_out  = (const float*)d_in[11];
    float* out = (float*)d_out;

    char* ws = (char*)d_ws;
    unsigned short* xn    = (unsigned short*)(ws + OFF_XN);
    unsigned short* wtin  = (unsigned short*)(ws + OFF_WTIN);
    unsigned short* xb    = (unsigned short*)(ws + OFF_XB);
    unsigned short* zb    = (unsigned short*)(ws + OFF_Z);
    unsigned short* ub    = (unsigned short*)(ws + OFF_U);
    unsigned short* wtx   = (unsigned short*)(ws + OFF_WTX);
    float*          dbc   = (float*)(ws + OFF_DBC);
    unsigned short* dtp   = (unsigned short*)(ws + OFF_DTP);
    unsigned short* wtdt  = (unsigned short*)(ws + OFF_WTDT);
    float*          dtf   = (float*)(ws + OFF_DTF);
    unsigned short* ym    = (unsigned short*)(ws + OFF_YM);
    unsigned short* wtout = (unsigned short*)(ws + OFF_WTOUT);

    // weight transposes -> bf16 [N][K]
    k_transpose_bf16<<<(DM * NXZ + 255) / 256, 256, 0, stream>>>(W_in, wtin, DM, NXZ);
    k_transpose_bf16<<<(DI * NDBC + 255) / 256, 256, 0, stream>>>(W_x, wtx, DI, NDBC);
    k_transpose_bf16<<<(DR * DI + 255) / 256, 256, 0, stream>>>(W_dt, wtdt, DR, DI);
    k_transpose_bf16<<<(DI * DM + 255) / 256, 256, 0, stream>>>(W_out, wtout, DI, DM);

    // layernorm -> xn (bf16)
    k_layernorm<<<NROW, 256, 0, stream>>>(x, ln_g, ln_b, xn);

    dim3 blk(32, 4);
    // GEMM1: xz = xn @ W_in, split-store xb | z (bf16)  [M=4096,N=4096,K=1024]
    k_gemm_wmma<0><<<dim3(NXZ / 32, NROW / 128), blk, 0, stream>>>(
        xn, DM, wtin, DM, DM, nullptr, xb, zb, DI, nullptr, DI);

    // conv + SiLU -> u (bf16)
    k_conv_silu<<<(int)(((size_t)NROW * DI + 255) / 256), 256, 0, stream>>>(xb, conv_w, conv_b, ub);

    // GEMM2: dbc = u @ W_x (f32)  [M=4096,N=96,K=2048]
    k_gemm_wmma<1><<<dim3(NDBC / 32, NROW / 128), blk, 0, stream>>>(
        ub, DI, wtx, DI, DI, dbc, nullptr, nullptr, 0, nullptr, NDBC);

    // dt-rank slice -> bf16
    k_dtp<<<(NROW * DR + 255) / 256, 256, 0, stream>>>(dbc, dtp);

    // GEMM3: dt = softplus(dtp @ W_dt + b_dt) (f32)  [M=4096,N=2048,K=64]
    k_gemm_wmma<2><<<dim3(DI / 32, NROW / 128), blk, 0, stream>>>(
        dtp, DR, wtdt, DR, DR, dtf, nullptr, nullptr, 0, b_dt, DI);

    // selective scan -> ym = (y + u*D) * silu(z) (bf16)
    k_scan<<<(BB * DI + 255) / 256, 256, 0, stream>>>(dtf, dbc, ub, zb, A_log, D_par, ym);

    // GEMM4: out = ym @ W_out + x (f32)  [M=4096,N=1024,K=2048]
    k_gemm_wmma<3><<<dim3(DM / 32, NROW / 128), blk, 0, stream>>>(
        ym, DI, wtout, DI, DI, out, nullptr, nullptr, 0, x, DM);
}
